// CrossModalAttention_62027917689485
// MI455X (gfx1250) — matrix operand order repaired and checked by reading, not compiled, not run on hardware
//
#include <hip/hip_runtime.h>
#include <hip/hip_bf16.h>

// Problem constants (match reference)
#define IN_DIM 512
#define QKD    64
#define NBATCH 8
#define NPIX   4096
#define CHUNK  32   // keys per streamed chunk in attention

typedef __attribute__((ext_vector_type(16))) __bf16 v16bf;
typedef __attribute__((ext_vector_type(8)))  float  v8f;
typedef __attribute__((ext_vector_type(4)))  unsigned v4u;
typedef __attribute__((ext_vector_type(8)))  int      v8i;
typedef __attribute__((ext_vector_type(4)))  int      v4i;

struct alignas(32) Pack32 { uint4 a, b; };

__device__ __forceinline__ unsigned short f2bf(float f) {
  __bf16 h = (__bf16)f;
  return __builtin_bit_cast(unsigned short, h);
}

// B-fragment: 16 bf16 contiguous halves per lane (32B): B[k=16h+2v+t][col]
__device__ __forceinline__ v16bf ldB(const unsigned short* p) {
  Pack32 t{((const uint4*)p)[0], ((const uint4*)p)[1]};
  return __builtin_bit_cast(v16bf, t);
}
// A-fragment: two 16B chunks at +0 and +8 halves (K = 16h..16h+15)
__device__ __forceinline__ v16bf ldA(const unsigned short* p) {
  Pack32 t{*(const uint4*)p, *(const uint4*)(p + 8)};
  return __builtin_bit_cast(v16bf, t);
}

__device__ __forceinline__ v8f wmma_bf16(v16bf a, v16bf b, v8f c) {
  return __builtin_amdgcn_wmma_f32_16x16x32_bf16(false, a, false, b, (short)0, c,
                                                 false, false);
}

__device__ __forceinline__ float hmax16(float v) {
  v = fmaxf(v, __shfl_xor(v, 1, 32));
  v = fmaxf(v, __shfl_xor(v, 2, 32));
  v = fmaxf(v, __shfl_xor(v, 4, 32));
  v = fmaxf(v, __shfl_xor(v, 8, 32));
  return v;
}
__device__ __forceinline__ float hsum16(float v) {
  v += __shfl_xor(v, 1, 32);
  v += __shfl_xor(v, 2, 32);
  v += __shfl_xor(v, 4, 32);
  v += __shfl_xor(v, 8, 32);
  return v;
}

// gfx1250 async memory->LDS DMA (ASYNCcnt-tracked).
__device__ __forceinline__ void async_b128_to_lds(void* lds_ptr, const void* gptr) {
  unsigned loff = (unsigned)(size_t)lds_ptr;
  asm volatile("global_load_async_to_lds_b128 %0, %1, off"
               :: "v"(loff), "v"(gptr) : "memory");
}
__device__ __forceinline__ void wait_async0() {
  asm volatile("s_wait_asynccnt 0" ::: "memory");
}

// Tensor Data Mover: one descriptor moves the whole V chunk
// (512 rows x 64B, row stride 8KB) from global to LDS. D# per ISA 8.3-8.5.
__device__ __forceinline__ void tdm_load_v_chunk(unsigned lds_addr, const void* gaddr) {
  unsigned long long ga = (unsigned long long)gaddr;
  v4u g0 = {1u,                                     // count=1, user mode
            lds_addr,                               // LDS byte address
            (unsigned)ga,                           // global_addr[31:0]
            ((unsigned)(ga >> 32) & 0x1FFFFFFu) | (2u << 30)};  // addr[56:32], type=2
  v8i g1 = {(int)(2u << 16),                 // workgroup_mask=0, data_size=4B
            (int)((2048u & 0xFFFFu) << 16),  // tensor_dim0 lo (2048 dwords per row)
            (int)((512u & 0xFFFFu) << 16),   // tensor_dim0 hi=0, tensor_dim1=512
            (int)(16u << 16),                // tensor_dim1 hi=0, tile_dim0=16 dwords
            (int)512,                        // tile_dim1=512, tile_dim2=0
            (int)2048,                       // tensor_dim0_stride lo (dwords)
            0, 0};                           // stride hi / dim1_stride = 0
  v4i g2 = {0, 0, 0, 0};
  v4i g3 = {0, 0, 0, 0};
#if defined(__clang_major__) && (__clang_major__ >= 23)
  v8i g4 = {0, 0, 0, 0, 0, 0, 0, 0};
  __builtin_amdgcn_tensor_load_to_lds(g0, g1, g2, g3, g4, 0);
#else
  __builtin_amdgcn_tensor_load_to_lds(g0, g1, g2, g3, 0);
#endif
}

// ---------------------------------------------------------------------------
__global__ __launch_bounds__(256) void convert_bf16_kernel(
    const float* __restrict__ s, unsigned short* __restrict__ d, int n) {
  int i = blockIdx.x * 256 + threadIdx.x;
  if (i < n) d[i] = f2bf(s[i]);
}

// ---------------------------------------------------------------------------
// Tiled transpose+convert: src f32 [B][C][N] -> dstT bf16 [B][N][C]
// ---------------------------------------------------------------------------
__global__ __launch_bounds__(256) void transpose_bf16_kernel(
    const float* __restrict__ src, unsigned short* __restrict__ dstT) {
  __shared__ float T[64][65];
  const int tid = threadIdx.x;
  const int n0 = blockIdx.x * 64;
  const int c0 = blockIdx.y * 64;
  const int bb = blockIdx.z;
#pragma unroll
  for (int it = 0; it < 4; ++it) {
    const int c = it * 16 + (tid >> 4);
    const int n = (tid & 15) * 4;
    const float4 v =
        *(const float4*)(src + ((size_t)(bb * IN_DIM + c0 + c)) * NPIX + n0 + n);
    T[c][n + 0] = v.x; T[c][n + 1] = v.y; T[c][n + 2] = v.z; T[c][n + 3] = v.w;
  }
  __syncthreads();
#pragma unroll
  for (int it = 0; it < 2; ++it) {
    const int n = it * 32 + (tid >> 3);
    const int c = (tid & 7) * 8;
    alignas(16) unsigned short tmp[8];
#pragma unroll
    for (int j = 0; j < 8; ++j) tmp[j] = f2bf(T[c + j][n]);
    *(uint4*)(dstT + ((size_t)(bb * NPIX + n0 + n)) * IN_DIM + c0 + c) =
        *(const uint4*)tmp;
  }
}

// ---------------------------------------------------------------------------
// Q/K projection WMMA GEMM, software-pipelined (prefetch chunk k+1 into regs
// while chunk k's WMMAs issue -> partial loadcnt waits instead of wait 0).
// grid (N/128, 1, B); wave w -> 16 n-rows, all 64 o.
// ---------------------------------------------------------------------------
__global__ __launch_bounds__(256) void gemm_qk_kernel(
    const unsigned short* __restrict__ aT, const unsigned short* __restrict__ Wbf,
    const float* __restrict__ bias, unsigned short* __restrict__ dst) {
  const int tid = threadIdx.x, w = tid >> 5, lane = tid & 31;
  const int h = lane >> 4, col = lane & 15;
  const int bb = blockIdx.z;
  const int n0 = blockIdx.x * 128 + w * 16;

  const unsigned short* ap = aT + ((size_t)(bb * NPIX + n0 + col)) * IN_DIM + h * 16;
  const unsigned short* bp = Wbf + (size_t)col * IN_DIM + h * 16;

  v8f acc[4];
  const v8f vzero = {0, 0, 0, 0, 0, 0, 0, 0};
#pragma unroll
  for (int nt = 0; nt < 4; ++nt) acc[nt] = vzero;

  v16bf af  = ldA(ap);
  v16bf bf0 = ldB(bp);
  v16bf bf1 = ldB(bp + 16 * IN_DIM);
  v16bf bf2 = ldB(bp + 32 * IN_DIM);
  v16bf bf3 = ldB(bp + 48 * IN_DIM);
#pragma unroll 4
  for (int kc = 0; kc < IN_DIM / 32; ++kc) {
    const int c1 = (kc + 1) * 32;
    v16bf afn = af, b0n = bf0, b1n = bf1, b2n = bf2, b3n = bf3;
    if (kc + 1 < IN_DIM / 32) {
      afn = ldA(ap + c1);
      b0n = ldB(bp + c1);
      b1n = ldB(bp + 16 * IN_DIM + c1);
      b2n = ldB(bp + 32 * IN_DIM + c1);
      b3n = ldB(bp + 48 * IN_DIM + c1);
    }
    acc[0] = wmma_bf16(af, bf0, acc[0]);
    acc[1] = wmma_bf16(af, bf1, acc[1]);
    acc[2] = wmma_bf16(af, bf2, acc[2]);
    acc[3] = wmma_bf16(af, bf3, acc[3]);
    af = afn; bf0 = b0n; bf1 = b1n; bf2 = b2n; bf3 = b3n;
  }
#pragma unroll
  for (int nt = 0; nt < 4; ++nt) {
    const int o = nt * 16 + col;
    const float bo = bias[o];
#pragma unroll
    for (int r = 0; r < 8; ++r)
      dst[((size_t)(bb * NPIX + n0 + r + 8 * h)) * QKD + o] = f2bf(acc[nt][r] + bo);
  }
}

// ---------------------------------------------------------------------------
// V projection WMMA GEMM, software-pipelined. grid (N/64, C/128, B).
// ---------------------------------------------------------------------------
__global__ __launch_bounds__(256) void gemm_v_kernel(
    const unsigned short* __restrict__ zT, const unsigned short* __restrict__ Wvbf,
    const float* __restrict__ bv, unsigned short* __restrict__ Vbf) {
  const int tid = threadIdx.x, w = tid >> 5, lane = tid & 31;
  const int h = lane >> 4, col = lane & 15;
  const int bb = blockIdx.z;
  const int n0 = blockIdx.x * 64;
  const int o0 = blockIdx.y * 128 + w * 16;

  const unsigned short* ap = Wvbf + ((size_t)(o0 + col)) * IN_DIM + h * 16;
  const unsigned short* bp = zT + ((size_t)(bb * NPIX + n0 + col)) * IN_DIM + h * 16;

  v8f acc[4];
  const v8f vzero = {0, 0, 0, 0, 0, 0, 0, 0};
#pragma unroll
  for (int nt = 0; nt < 4; ++nt) acc[nt] = vzero;

  v16bf af  = ldA(ap);
  v16bf bf0 = ldB(bp);
  v16bf bf1 = ldB(bp + 16 * IN_DIM);
  v16bf bf2 = ldB(bp + 32 * IN_DIM);
  v16bf bf3 = ldB(bp + 48 * IN_DIM);
#pragma unroll 4
  for (int kc = 0; kc < IN_DIM / 32; ++kc) {
    const int c1 = (kc + 1) * 32;
    v16bf afn = af, b0n = bf0, b1n = bf1, b2n = bf2, b3n = bf3;
    if (kc + 1 < IN_DIM / 32) {
      afn = ldA(ap + c1);
      b0n = ldB(bp + c1);
      b1n = ldB(bp + 16 * IN_DIM + c1);
      b2n = ldB(bp + 32 * IN_DIM + c1);
      b3n = ldB(bp + 48 * IN_DIM + c1);
    }
    acc[0] = wmma_bf16(af, bf0, acc[0]);
    acc[1] = wmma_bf16(af, bf1, acc[1]);
    acc[2] = wmma_bf16(af, bf2, acc[2]);
    acc[3] = wmma_bf16(af, bf3, acc[3]);
    af = afn; bf0 = b0n; bf1 = b1n; bf2 = b2n; bf3 = b3n;
  }
  float bv8[8];
#pragma unroll
  for (int r = 0; r < 8; ++r) bv8[r] = bv[o0 + r + 8 * h];
#pragma unroll
  for (int nt = 0; nt < 4; ++nt)
#pragma unroll
    for (int r = 0; r < 8; ++r)
      Vbf[((size_t)(bb * IN_DIM + o0 + r + 8 * h)) * NPIX + n0 + nt * 16 + col] =
          f2bf(acc[nt][r] + bv8[r]);
}

// ---------------------------------------------------------------------------
// Fused flash attention + residual. Block = 8 wave32 = one 32-query tile.
// K chunk staged via per-lane async LDS DMA (ASYNCcnt); V chunk staged via a
// single Tensor Data Mover descriptor (TENSORcnt) issued by wave 0.
// ---------------------------------------------------------------------------
__global__ __launch_bounds__(256) void attn_kernel(
    const unsigned short* __restrict__ Qbf,
    const unsigned short* __restrict__ Kbf,
    const unsigned short* __restrict__ Vbf,
    const float* __restrict__ x, const float* __restrict__ gamma,
    float* __restrict__ out) {
  __shared__ __align__(32) unsigned short Ksh[CHUNK * QKD];     // [key][d]   4 KB
  __shared__ __align__(32) unsigned short Vsh[IN_DIM * CHUNK];  // [ch][key] 32 KB
  __shared__ __align__(32) unsigned short Psh[2][16 * CHUNK];   // 2 KB
  __shared__ float alphaSh[2][16];
  __shared__ float lSh[2][16];

  const int tid  = threadIdx.x;
  const int w    = tid >> 5;
  const int lane = tid & 31;
  const int h    = lane >> 4;
  const int col  = lane & 15;
  const int b    = blockIdx.x >> 7;
  const int qt   = blockIdx.x & 127;
  const int rowHalf = w >> 2;
  const int chBase  = (w & 3) * 128;
  const bool isEnergy = (w & 3) == 0;
  const int qrow0 = qt * 32 + rowHalf * 16;

  const unsigned short* qp =
      Qbf + ((size_t)(b * NPIX + qrow0 + col)) * QKD + h * 16;
  const v16bf qf0 = ldA(qp);
  const v16bf qf1 = ldA(qp + 32);

  float mrow[8], lrow[8];
#pragma unroll
  for (int r = 0; r < 8; ++r) { mrow[r] = -3.0e38f; lrow[r] = 0.f; }
  v8f acc[8];
  const v8f vzero = {0, 0, 0, 0, 0, 0, 0, 0};
#pragma unroll
  for (int nt = 0; nt < 8; ++nt) acc[nt] = vzero;

  for (int jc = 0; jc < NPIX / CHUNK; ++jc) {
    const int jb = jc * CHUNK;
    __syncthreads();  // previous chunk's compute done before LDS overwrite
    {  // K chunk: per-lane async DMA, 32 keys x 128B rows, 8 lanes per row
      const int row = tid >> 3, seg = tid & 7;
      async_b128_to_lds(Ksh + row * QKD + seg * 8,
                        Kbf + ((size_t)(b * NPIX + jb + row)) * QKD + seg * 8);
    }
    if (tid < 32) {  // V chunk: one TDM descriptor (512 rows x 64B, stride 8KB)
      tdm_load_v_chunk((unsigned)(size_t)Vsh,
                       Vbf + (size_t)b * IN_DIM * NPIX + jb);
      __builtin_amdgcn_s_wait_tensorcnt(0);
    }
    if (jc + 1 < NPIX / CHUNK) {
      __builtin_prefetch(Kbf + ((size_t)(b * NPIX + jb + CHUNK + (tid >> 3))) * QKD, 0, 1);
      __builtin_prefetch(Vbf + ((size_t)(b * IN_DIM + (tid >> 2))) * NPIX + jb + CHUNK, 0, 1);
    }
    wait_async0();
    __syncthreads();  // staged data visible block-wide

    if (isEnergy) {
      const unsigned short* kb = Ksh + col * QKD + h * 16;
      v8f e0 = vzero, e1 = vzero;
      e0 = wmma_bf16(qf0, ldB(kb), e0);
      e0 = wmma_bf16(qf1, ldB(kb + 32), e0);
      e1 = wmma_bf16(qf0, ldB(kb + 16 * QKD), e1);
      e1 = wmma_bf16(qf1, ldB(kb + 16 * QKD + 32), e1);
#pragma unroll
      for (int r = 0; r < 8; ++r) {
        float mx   = hmax16(fmaxf(e0[r], e1[r]));
        float mnew = fmaxf(mrow[r], mx);
        float a    = __expf(mrow[r] - mnew);
        float p0   = __expf(e0[r] - mnew);
        float p1   = __expf(e1[r] - mnew);
        float s    = hsum16(p0 + p1);
        lrow[r] = a * lrow[r] + s;
        mrow[r] = mnew;
        Psh[rowHalf][(r + 8 * h) * CHUNK + col]      = f2bf(p0);
        Psh[rowHalf][(r + 8 * h) * CHUNK + 16 + col] = f2bf(p1);
        if (col == r) alphaSh[rowHalf][8 * h + r] = a;
      }
    }
    __syncthreads();  // P and alpha published

    float alv[8];
#pragma unroll
    for (int r = 0; r < 8; ++r) alv[r] = alphaSh[rowHalf][8 * h + r];
#pragma unroll
    for (int nt = 0; nt < 8; ++nt)
#pragma unroll
      for (int r = 0; r < 8; ++r) acc[nt][r] *= alv[r];

    const v16bf pf = ldA(Psh[rowHalf] + col * CHUNK + h * 16);
#pragma unroll
    for (int nt = 0; nt < 8; ++nt) {
      const unsigned short* vb = Vsh + (chBase + nt * 16 + col) * CHUNK + h * 16;
      acc[nt] = wmma_bf16(pf, ldB(vb), acc[nt]);
    }
  }

  __syncthreads();
  if (isEnergy) {
#pragma unroll
    for (int r = 0; r < 8; ++r)
      if (col == r) lSh[rowHalf][8 * h + r] = lrow[r];
  }
  __syncthreads();

  float lv[8];
#pragma unroll
  for (int r = 0; r < 8; ++r) lv[r] = 1.0f / lSh[rowHalf][8 * h + r];
  const float g = gamma[0];
#pragma unroll
  for (int nt = 0; nt < 8; ++nt) {
    const int ch = chBase + nt * 16 + col;
    const size_t base = ((size_t)(b * IN_DIM + ch)) * NPIX + (size_t)(qrow0 + 8 * h);
    float4 x0 = *(const float4*)(x + base);
    float4 x1 = *(const float4*)(x + base + 4);
    float4 o0, o1;
    o0.x = g * (acc[nt][0] * lv[0]) + x0.x;
    o0.y = g * (acc[nt][1] * lv[1]) + x0.y;
    o0.z = g * (acc[nt][2] * lv[2]) + x0.z;
    o0.w = g * (acc[nt][3] * lv[3]) + x0.w;
    o1.x = g * (acc[nt][4] * lv[4]) + x1.x;
    o1.y = g * (acc[nt][5] * lv[5]) + x1.y;
    o1.z = g * (acc[nt][6] * lv[6]) + x1.z;
    o1.w = g * (acc[nt][7] * lv[7]) + x1.w;
    *(float4*)(out + base)     = o0;
    *(float4*)(out + base + 4) = o1;
  }
}

// ---------------------------------------------------------------------------
extern "C" void kernel_launch(void* const* d_in, const int* in_sizes, int n_in,
                              void* d_out, int out_size, void* d_ws, size_t ws_size,
                              hipStream_t stream) {
  (void)in_sizes; (void)n_in; (void)out_size; (void)ws_size;
  const float* x     = (const float*)d_in[0];
  const float* y     = (const float*)d_in[1];
  const float* z     = (const float*)d_in[2];
  const float* Wq    = (const float*)d_in[3];
  const float* bq    = (const float*)d_in[4];
  const float* Wk    = (const float*)d_in[5];
  const float* bk    = (const float*)d_in[6];
  const float* Wv    = (const float*)d_in[7];
  const float* bv    = (const float*)d_in[8];
  const float* gamma = (const float*)d_in[9];
  float* out = (float*)d_out;

  char* ws = (char*)d_ws;
  unsigned short* T0   = (unsigned short*)ws;                            // 32 MB
  unsigned short* Qbf  = (unsigned short*)(ws + ((size_t)32 << 20));     //  4 MB
  unsigned short* Kbf  = (unsigned short*)(ws + ((size_t)36 << 20));     //  4 MB
  unsigned short* Vbf  = (unsigned short*)(ws + ((size_t)40 << 20));     // 32 MB
  unsigned short* Wqbf = (unsigned short*)(ws + ((size_t)72 << 20));     // 64 KB
  unsigned short* Wkbf = (unsigned short*)(ws + ((size_t)72 << 20) + (64 << 10));
  unsigned short* Wvbf = (unsigned short*)(ws + ((size_t)72 << 20) + (128 << 10));

  convert_bf16_kernel<<<dim3((QKD * IN_DIM + 255) / 256), 256, 0, stream>>>(
      Wq, Wqbf, QKD * IN_DIM);
  convert_bf16_kernel<<<dim3((QKD * IN_DIM + 255) / 256), 256, 0, stream>>>(
      Wk, Wkbf, QKD * IN_DIM);
  convert_bf16_kernel<<<dim3((IN_DIM * IN_DIM + 255) / 256), 256, 0, stream>>>(
      Wv, Wvbf, IN_DIM * IN_DIM);

  dim3 tg(NPIX / 64, IN_DIM / 64, NBATCH);
  transpose_bf16_kernel<<<tg, 256, 0, stream>>>(x, T0);
  gemm_qk_kernel<<<dim3(NPIX / 128, 1, NBATCH), 256, 0, stream>>>(T0, Wqbf, bq, Qbf);
  transpose_bf16_kernel<<<tg, 256, 0, stream>>>(y, T0);
  gemm_qk_kernel<<<dim3(NPIX / 128, 1, NBATCH), 256, 0, stream>>>(T0, Wkbf, bk, Kbf);
  transpose_bf16_kernel<<<tg, 256, 0, stream>>>(z, T0);
  gemm_v_kernel<<<dim3(NPIX / 64, IN_DIM / 128, NBATCH), 256, 0, stream>>>(
      T0, Wvbf, bv, Vbf);

  attn_kernel<<<dim3(NBATCH * (NPIX / 32)), 256, 0, stream>>>(
      Qbf, Kbf, Vbf, x, gamma, out);
}
